// CMRGroupLayer_39187281609208
// MI455X (gfx1250) — matrix operand order repaired and checked by reading, not compiled, not run on hardware
//
#include <hip/hip_runtime.h>
#include <stdint.h>

#define Bdim 8
#define Sdim 1024
#define Ddim 1024
#define Fdim 4096
#define Edim 8
#define Ttok (Bdim * Sdim) // 8192 tokens

typedef __attribute__((ext_vector_type(16))) __bf16   v16bf;
typedef __attribute__((ext_vector_type(8)))  float    v8f;
typedef __attribute__((ext_vector_type(4)))  uint32_t v4u;
typedef __attribute__((ext_vector_type(4)))  int      v4i;
typedef __attribute__((ext_vector_type(4)))  float    v4f;

union FragAB { v4u q[2]; v16bf v; };

// CDNA5 async global->LDS path (ASYNCcnt-tracked), if the toolchain declares it.
#if defined(__has_builtin)
#if __has_builtin(__builtin_amdgcn_global_load_async_to_lds_b128) && \
    __has_builtin(__builtin_amdgcn_s_wait_asynccnt)
#define USE_ASYNC_LDS 1
#endif
#endif

#ifdef USE_ASYNC_LDS
typedef __attribute__((address_space(1))) v4i* g_v4i_p; // global (AS1) src (param is non-const)
typedef __attribute__((address_space(3))) v4i* l_v4i_p; // LDS (AS3) dst
#endif

__device__ __forceinline__ unsigned short f2bf(float f) {
  uint32_t u = __float_as_uint(f);
  u += 0x7FFFu + ((u >> 16) & 1u); // round-to-nearest-even
  return (unsigned short)(u >> 16);
}

// ---------------- small utility kernels ----------------

__global__ void zero_small(int* cnt, float* ce, float* scal) {
  int t = threadIdx.x;
  if (t < Edim) { cnt[t] = 0; ce[t] = 0.f; }
  if (t == Edim) { scal[0] = 0.f; scal[1] = 0.f; }
}

__global__ __launch_bounds__(256) void cvt_f32_bf16(const float* __restrict__ src,
                                                    unsigned short* __restrict__ dst, int n) {
  int i = (blockIdx.x * 256 + threadIdx.x) * 4;
  if (i >= n) return;
  v4f f = *(const v4f*)(src + i);
  uint32_t o0 = (uint32_t)f2bf(f.x) | ((uint32_t)f2bf(f.y) << 16);
  uint32_t o1 = (uint32_t)f2bf(f.z) | ((uint32_t)f2bf(f.w) << 16);
  uint32_t* d = (uint32_t*)(dst + i);
  d[0] = o0; d[1] = o1;
}

// dst[c*R + r] = bf16(src[r*C + c]); grid (C/32, R/32, nmat), block (32,8)
__global__ __launch_bounds__(256) void transpose_cvt(const float* __restrict__ srcB,
                                                     unsigned short* __restrict__ dstB,
                                                     int R, int C) {
  __shared__ float tile[32][33];
  const float* src = srcB + (size_t)blockIdx.z * R * C;
  unsigned short* dst = dstB + (size_t)blockIdx.z * R * C;
  int c0 = blockIdx.x * 32, r0 = blockIdx.y * 32;
  int tx = threadIdx.x, ty = threadIdx.y;
#pragma unroll
  for (int i = 0; i < 32; i += 8)
    tile[ty + i][tx] = src[(size_t)(r0 + ty + i) * C + (c0 + tx)];
  __syncthreads();
#pragma unroll
  for (int i = 0; i < 32; i += 8)
    dst[(size_t)(c0 + ty + i) * R + (r0 + tx)] = f2bf(tile[tx][ty + i]);
}

// ---------------- router / gate: one wave per token ----------------

__global__ __launch_bounds__(256) void router_kernel(
    const float* __restrict__ x, const float* __restrict__ wg_w, const float* __restrict__ wg_b,
    const float* __restrict__ Wr, const unsigned char* __restrict__ pad,
    float* __restrict__ gates, float* __restrict__ mscale,
    int* __restrict__ cnt, int* __restrict__ idx,
    float* __restrict__ ce, float* __restrict__ scal /*[0]=used,[1]=nonpad*/) {
  __shared__ float sCE[Edim];
  __shared__ float sUsed, sNon;
  int tid = threadIdx.x;
  if (tid < Edim) sCE[tid] = 0.f;
  if (tid == Edim) sUsed = 0.f;
  if (tid == Edim + 1) sNon = 0.f;
  __syncthreads();

  int wave = tid >> 5, lane = tid & 31;
  int t = blockIdx.x * 8 + wave;
  const float* xr = x + (size_t)t * Ddim;
  float accg = 0.f;
  float accr[Edim] = {0.f, 0.f, 0.f, 0.f, 0.f, 0.f, 0.f, 0.f};
  for (int k = lane; k < Ddim; k += 32) {
    float xv = xr[k];
    accg += xv * wg_w[k];
    const v4f* wr = (const v4f*)(Wr + (size_t)k * Edim);
    v4f w0 = wr[0], w1 = wr[1];
    accr[0] += xv * w0.x; accr[1] += xv * w0.y; accr[2] += xv * w0.z; accr[3] += xv * w0.w;
    accr[4] += xv * w1.x; accr[5] += xv * w1.y; accr[6] += xv * w1.z; accr[7] += xv * w1.w;
  }
#pragma unroll
  for (int off = 16; off > 0; off >>= 1) {
    accg += __shfl_xor(accg, off, 32);
#pragma unroll
    for (int e = 0; e < Edim; e++) accr[e] += __shfl_xor(accr[e], off, 32);
  }
  if (lane == 0) {
    float np = pad[t] ? 0.f : 1.f;
    float g = 1.f / (1.f + __expf(-(accg + wg_b[0])));
    g *= np;
    float m = accr[0]; int tp = 0;
#pragma unroll
    for (int e = 1; e < Edim; e++) if (accr[e] > m) { m = accr[e]; tp = e; }
    float p[Edim]; float s = 0.f;
#pragma unroll
    for (int e = 0; e < Edim; e++) { p[e] = __expf(accr[e] - m); s += p[e]; }
    float inv = 1.f / s;
    gates[t] = g;
    mscale[t] = g * p[tp] * inv;              // gate * top1 prob
    int pos = atomicAdd(&cnt[tp], 1);
    idx[tp * Ttok + pos] = t;
#pragma unroll
    for (int e = 0; e < Edim; e++) atomicAdd(&sCE[e], p[e] * inv);
    atomicAdd(&sUsed, g);
    atomicAdd(&sNon, np);
  }
  __syncthreads();
  if (tid < Edim) atomicAdd(&ce[tid], sCE[tid]);
  if (tid == Edim) atomicAdd(&scal[0], sUsed);
  if (tid == Edim + 1) atomicAdd(&scal[1], sNon);
}

// ---------------- WMMA GEMM (double-buffered LDS, 128x128 block tile) ----------------
// MODE 0: H = relu(A*B + bias) bf16   (shared FFN up)
// MODE 1: same, rows gathered by idx  (MoE up, per expert = blockIdx.z)
// MODE 2: out = (A*B + bias) * (1 - gates[row])      (shared FFN down)
// MODE 3: out += mscale[tok] * (A*B + bias), gathered (MoE down)
template <int MODE>
__global__ __launch_bounds__(256) void gemm_wmma(
    const unsigned short* __restrict__ Abase, int lda,
    const unsigned short* __restrict__ Bbase,   // [N][K], per-expert stride Ntot*K
    const float* __restrict__ biasBase,         // per-expert stride Ntot
    float* __restrict__ OutF, unsigned short* __restrict__ OutH,
    int ldc, int K, int Ntot,
    const int* __restrict__ idxBase, const int* __restrict__ cntp,
    const float* __restrict__ gates, const float* __restrict__ mscale) {
  constexpr bool GATHER = (MODE == 1 || MODE == 3);
  constexpr bool STOREH = (MODE <= 1);

  __shared__ unsigned short As[2][128 * 40]; // 128 rows x 32 K, stride 40, ping-pong
  __shared__ unsigned short Bs[2][128 * 40]; // 128 cols x 32 K, stride 40, ping-pong
  __shared__ int IdxS[128];

  int e = blockIdx.z;
  const unsigned short* Bm = Bbase + (size_t)e * Ntot * K;
  const float* bias = biasBase + (size_t)e * Ntot;
  int m0 = blockIdx.y * 128;
  int n0 = blockIdx.x * 128;
  int Mv = Ttok;
  if (GATHER) {
    Mv = cntp[e];
    if (m0 >= Mv) return;
  }
  int rowsValid = Mv - m0; if (rowsValid > 128) rowsValid = 128;

  int tid = threadIdx.x;
  if (GATHER) {
    if (tid < 128) {
      int rr = tid < rowsValid ? tid : (rowsValid - 1);
      IdxS[tid] = idxBase[e * Ttok + m0 + rr];
    }
    __syncthreads(); // IdxS visible before first staging pass
  }

  int wave = tid >> 5, lane = tid & 31;
  int wm = wave >> 1, wn = wave & 1;   // 4(M) x 2(N) waves; wave tile 32 x 64
  int half = lane >> 4, l16 = lane & 15;

  v8f acc[2][4];
#pragma unroll
  for (int mi = 0; mi < 2; mi++)
#pragma unroll
    for (int ni = 0; ni < 4; ni++)
      acc[mi][ni] = (v8f){0.f, 0.f, 0.f, 0.f, 0.f, 0.f, 0.f, 0.f};

  int cst = (tid & 3) * 8;   // K-chunk (8 bf16 = 16B)
  int rst = tid >> 2;        // 0..63 (two passes cover 128 rows)

  // per-thread global source addresses for one K-stage
  auto aSrc = [&](int p, int k0) -> const unsigned short* {
    int r = rst + p * 64;
    int rt = GATHER ? IdxS[r] : (m0 + r);
    return Abase + (size_t)rt * lda + k0 + cst;
  };
  auto bSrc = [&](int p, int k0) -> const unsigned short* {
    return Bm + (size_t)(n0 + rst + p * 64) * K + k0 + cst;
  };

#ifdef USE_ASYNC_LDS
  auto stage_async = [&](int buf, int k0) {
#pragma unroll
    for (int p = 0; p < 2; p++) {
      __builtin_amdgcn_global_load_async_to_lds_b128(
          (g_v4i_p)aSrc(p, k0),
          (l_v4i_p)&As[buf][(rst + p * 64) * 40 + cst], 0, 0);
      __builtin_amdgcn_global_load_async_to_lds_b128(
          (g_v4i_p)bSrc(p, k0),
          (l_v4i_p)&Bs[buf][(rst + p * 64) * 40 + cst], 0, 0);
    }
  };
#else
  v4u ra[2], rb[2];
  auto stage_load = [&](int k0) {
#pragma unroll
    for (int p = 0; p < 2; p++) {
      ra[p] = *(const v4u*)aSrc(p, k0);
      rb[p] = *(const v4u*)bSrc(p, k0);
    }
  };
  auto stage_store = [&](int buf) {
#pragma unroll
    for (int p = 0; p < 2; p++) {
      *(v4u*)(&As[buf][(rst + p * 64) * 40 + cst]) = ra[p];
      *(v4u*)(&Bs[buf][(rst + p * 64) * 40 + cst]) = rb[p];
    }
  };
#endif

  int nstage = K >> 5;

  // prologue: fill buffer 0
#ifdef USE_ASYNC_LDS
  stage_async(0, 0);
  __builtin_amdgcn_s_wait_asynccnt(0);
#else
  stage_load(0);
  stage_store(0);
#endif
  __syncthreads();

  for (int s = 0; s < nstage; s++) {
    int cur = s & 1;
    bool more = (s + 1) < nstage;
#ifdef USE_ASYNC_LDS
    if (more) stage_async(cur ^ 1, (s + 1) << 5); // in flight during compute
#else
    if (more) stage_load((s + 1) << 5);           // in flight during compute
#endif

    FragAB a[2], b[4];
#pragma unroll
    for (int mi = 0; mi < 2; mi++) {
      int r = wm * 32 + mi * 16 + l16;
      a[mi].q[0] = *(const v4u*)(&As[cur][r * 40 + half * 8]);       // K = h*8 .. h*8+7
      a[mi].q[1] = *(const v4u*)(&As[cur][r * 40 + 16 + half * 8]);  // K = 16+h*8 ..
    }
#pragma unroll
    for (int ni = 0; ni < 4; ni++) {
      int rn = wn * 64 + ni * 16 + l16;
      b[ni].q[0] = *(const v4u*)(&Bs[cur][rn * 40 + half * 8]);       // K = h*8 .. h*8+7
      b[ni].q[1] = *(const v4u*)(&Bs[cur][rn * 40 + 16 + half * 8]);  // K = 16+h*8 ..
    }
#pragma unroll
    for (int mi = 0; mi < 2; mi++)
#pragma unroll
      for (int ni = 0; ni < 4; ni++)
        acc[mi][ni] = __builtin_amdgcn_wmma_f32_16x16x32_bf16(
            false, a[mi].v, false, b[ni].v, (short)0, acc[mi][ni], false, false);

#ifdef USE_ASYNC_LDS
    if (more) __builtin_amdgcn_s_wait_asynccnt(0);
#else
    if (more) stage_store(cur ^ 1);
#endif
    __syncthreads(); // next buffer ready; current buffer free for overwrite next iter
  }

  // epilogue: lane holds rows (wm*32 + mi*16 + half*8 + i), col (n0 + wn*64 + ni*16 + l16)
#pragma unroll
  for (int mi = 0; mi < 2; mi++) {
#pragma unroll
    for (int ni = 0; ni < 4; ni++) {
      int ncol = n0 + wn * 64 + ni * 16 + l16;
      float bval = bias[ncol];
#pragma unroll
      for (int i = 0; i < 8; i++) {
        int r = wm * 32 + mi * 16 + half * 8 + i;
        if (!GATHER || r < rowsValid) {
          float v = acc[mi][ni][i] + bval;
          if (STOREH) {
            v = v > 0.f ? v : 0.f;
            int rt = GATHER ? IdxS[r] : (m0 + r);
            OutH[(size_t)rt * ldc + ncol] = f2bf(v);
          } else if (MODE == 2) {
            int t = m0 + r;
            OutF[(size_t)t * ldc + ncol] = v * (1.f - gates[t]);
          } else { // MODE 3: one expert owns each token -> race-free RMW
            int t = IdxS[r];
            size_t o = (size_t)t * ldc + ncol;
            OutF[o] += mscale[t] * v;
          }
        }
      }
    }
  }
}

__global__ void finalize(const int* cnt, const float* ce, const float* scal, float* out) {
  float inv = 1.f / (float)Ttok;
  float s = 0.f;
  for (int e = 0; e < Edim; e++) s += ((float)cnt[e] * inv) * (ce[e] * inv);
  size_t base = (size_t)Ttok * Ddim;
  out[base + 0] = (float)Edim * s;
  out[base + 1] = scal[0];
  out[base + 2] = scal[1];
}

// ---------------- launcher ----------------

extern "C" void kernel_launch(void* const* d_in, const int* in_sizes, int n_in,
                              void* d_out, int out_size, void* d_ws, size_t ws_size,
                              hipStream_t stream) {
  const float* x    = (const float*)d_in[0];
  const unsigned char* pad = (const unsigned char*)d_in[2];
  const float* wg_w = (const float*)d_in[3];
  const float* wg_b = (const float*)d_in[4];
  const float* Wr   = (const float*)d_in[5];
  const float* W1   = (const float*)d_in[6];
  const float* B1   = (const float*)d_in[7];
  const float* W2   = (const float*)d_in[8];
  const float* B2   = (const float*)d_in[9];
  const float* Wf1  = (const float*)d_in[10];
  const float* bf1  = (const float*)d_in[11];
  const float* Wf2  = (const float*)d_in[12];
  const float* bf2  = (const float*)d_in[13];
  float* out = (float*)d_out;

  char* ws = (char*)d_ws;
  size_t off = 0;
  auto alloc = [&](size_t bytes) -> char* {
    char* p = ws + off;
    off += (bytes + 255) & ~(size_t)255;
    return p;
  };
  unsigned short* xb    = (unsigned short*)alloc((size_t)Ttok * Ddim * 2);
  unsigned short* Wf1T  = (unsigned short*)alloc((size_t)Ddim * Fdim * 2);
  unsigned short* Wf2T  = (unsigned short*)alloc((size_t)Fdim * Ddim * 2);
  unsigned short* W1T   = (unsigned short*)alloc((size_t)Edim * Ddim * Fdim * 2);
  unsigned short* W2T   = (unsigned short*)alloc((size_t)Edim * Fdim * Ddim * 2);
  unsigned short* Hbuf  = (unsigned short*)alloc((size_t)Ttok * Fdim * 2); // reused FFN->MoE
  float* gatesB  = (float*)alloc(Ttok * 4);
  float* mscaleB = (float*)alloc(Ttok * 4);
  int*   idxB    = (int*)alloc((size_t)Edim * Ttok * 4);
  int*   cntB    = (int*)alloc(Edim * 4);
  float* ceB     = (float*)alloc(Edim * 4);
  float* scalB   = (float*)alloc(2 * 4);

  zero_small<<<1, 64, 0, stream>>>(cntB, ceB, scalB);

  // fp32 -> bf16 conversions (x plain; weights transposed to [N][K])
  cvt_f32_bf16<<<(Ttok * Ddim / 4 + 255) / 256, 256, 0, stream>>>(x, xb, Ttok * Ddim);
  transpose_cvt<<<dim3(Fdim / 32, Ddim / 32, 1), dim3(32, 8), 0, stream>>>(Wf1, Wf1T, Ddim, Fdim);
  transpose_cvt<<<dim3(Ddim / 32, Fdim / 32, 1), dim3(32, 8), 0, stream>>>(Wf2, Wf2T, Fdim, Ddim);
  transpose_cvt<<<dim3(Fdim / 32, Ddim / 32, Edim), dim3(32, 8), 0, stream>>>(W1, W1T, Ddim, Fdim);
  transpose_cvt<<<dim3(Ddim / 32, Fdim / 32, Edim), dim3(32, 8), 0, stream>>>(W2, W2T, Fdim, Ddim);

  router_kernel<<<Ttok / 8, 256, 0, stream>>>(x, wg_w, wg_b, Wr, pad, gatesB, mscaleB,
                                              cntB, idxB, ceB, scalB);

  // shared FFN
  gemm_wmma<0><<<dim3(Fdim / 128, Ttok / 128, 1), 256, 0, stream>>>(
      xb, Ddim, Wf1T, bf1, nullptr, Hbuf, Fdim, Ddim, Fdim, nullptr, nullptr, nullptr, nullptr);
  gemm_wmma<2><<<dim3(Ddim / 128, Ttok / 128, 1), 256, 0, stream>>>(
      Hbuf, Fdim, Wf2T, bf2, out, nullptr, Ddim, Fdim, Ddim, nullptr, nullptr, gatesB, nullptr);

  // routed MoE (top-1); worst-case grid, early-exit beyond per-expert count
  gemm_wmma<1><<<dim3(Fdim / 128, Ttok / 128, Edim), 256, 0, stream>>>(
      xb, Ddim, W1T, B1, nullptr, Hbuf, Fdim, Ddim, Fdim, idxB, cntB, nullptr, nullptr);
  gemm_wmma<3><<<dim3(Ddim / 128, Ttok / 128, Edim), 256, 0, stream>>>(
      Hbuf, Fdim, W2T, B2, out, nullptr, Ddim, Fdim, Ddim, idxB, cntB, nullptr, mscaleB);

  finalize<<<1, 1, 0, stream>>>(cntB, ceB, scalB, out);
}
